// LearnableSparsityPredictor_24283745091777
// MI455X (gfx1250) — matrix operand order repaired and checked by reading, not compiled, not run on hardware
//
#include <hip/hip_runtime.h>
#include <hip/hip_bf16.h>

typedef __attribute__((ext_vector_type(16))) _Float16 v16h;
typedef __attribute__((ext_vector_type(8)))  _Float16 v8h;
typedef __attribute__((ext_vector_type(8)))  float    v8f;

#define BB 8
#define NN_ 256
#define DD 128
#define HH 8
#define DH 64
#define DQ 32
#define K1P 112            // W1 K padded 97 -> 112 (mult of 16, 32B-aligned rows)
#define NWAVES 4

// LDS weight region offsets (in halves)
#define OFF_WE 0
#define OFF_W1 2048        // 32*64
#define OFF_W2 9216        // + 64*112
#define OFF_W3 11264       // + 32*64
#define W_TOTAL 11776      // + 16*32

// ---------------- weight prep: f32 -> f16, pad/transpose-free ----------------
__global__ void prep_weights_kernel(const float* __restrict__ W_edge,
                                    const float* __restrict__ W1,
                                    const float* __restrict__ W2,
                                    const float* __restrict__ W3,
                                    _Float16* __restrict__ wsh) {
    const int tid = threadIdx.x;
    // W_edge[:, :64] -> [32][64]
    for (int t = tid; t < 2048; t += 256) {
        int n = t >> 6, k = t & 63;
        wsh[OFF_WE + t] = (_Float16)W_edge[n * DD + k];
    }
    // W1 [64][97] -> [64][112] zero-padded
    for (int t = tid; t < 64 * K1P; t += 256) {
        int n = t / K1P, k = t % K1P;
        wsh[OFF_W1 + t] = (_Float16)(k < 97 ? W1[n * 97 + k] : 0.0f);
    }
    // W2 [32][64]
    for (int t = tid; t < 2048; t += 256) {
        wsh[OFF_W2 + t] = (_Float16)W2[t];
    }
    // W3 [8][32] -> [16][32] zero-padded
    for (int t = tid; t < 512; t += 256) {
        int n = t >> 5, k = t & 31;
        wsh[OFF_W3 + t] = (_Float16)(n < 8 ? W3[n * DQ + k] : 0.0f);
    }
}

// ---------------- nproj[b,n,:] = node_features[b,n,:] @ W_node.T + b_node ----
__global__ void nproj_kernel(const float* __restrict__ nf,
                             const float* __restrict__ Wn,
                             const float* __restrict__ bn,
                             float* __restrict__ nproj) {
    const int row = blockIdx.x;       // B*N rows
    const int t = threadIdx.x;        // 64 outputs
    const float* x = nf + (size_t)row * DD;
    const float* w = Wn + (size_t)t * DD;
    float acc = bn[t];
#pragma unroll 4
    for (int k = 0; k < DD; ++k) acc += x[k] * w[k];
    nproj[(size_t)row * DH + t] = acc;
}

// ---------------- WMMA fragment loaders (wave32 layouts, from LDS) ----------
// A (16xK f16): lane L -> row M=L%16; VGPR v in [0..3]: K = 8*(L/16) + 2v{+1},
// v in [4..7]: K = 16 + 8*(L/16) + 2(v-4){+1}. Per-lane: two runs of 8 halves.
static __device__ inline v16h lds_load_A(const _Float16* base, int strideH,
                                         int m, int l16, int kb) {
    const _Float16* p = base + m * strideH + kb + 8 * l16;
    v8h lo = *(const v8h*)(p);
    v8h hi = *(const v8h*)(p + 16);
    return __builtin_shufflevector(lo, hi, 0, 1, 2, 3, 4, 5, 6, 7,
                                   8, 9, 10, 11, 12, 13, 14, 15);
}
// B (Kx16 f16), weights stored [N][K] row-major: lane L -> col N=L%16,
// K = 8*(L/16) + 0..7 and 16 + 8*(L/16) + 0..7 (matches A). Two 16B loads.
static __device__ inline v16h lds_load_B(const _Float16* base, int strideH,
                                         int nrow, int l16, int kb) {
    const _Float16* p = base + nrow * strideH + kb + 8 * l16;
    v8h lo = *(const v8h*)(p);
    v8h hi = *(const v8h*)(p + 16);
    return __builtin_shufflevector(lo, hi, 0, 1, 2, 3, 4, 5, 6, 7,
                                   8, 9, 10, 11, 12, 13, 14, 15);
}

static __device__ inline v8f wmma_f16(v16h a, v16h b, v8f c) {
    return __builtin_amdgcn_wmma_f32_16x16x32_f16(false, a, false, b,
                                                  (short)0, c, false, false);
}

// ---------------- main per-edge MLP kernel ----------------------------------
__global__ __launch_bounds__(32 * NWAVES) void
edge_mlp_kernel(const float* __restrict__ nproj,
                const float* __restrict__ cds,
                const unsigned char* __restrict__ mask,
                const _Float16* __restrict__ wsh,
                const float* __restrict__ b_edge,
                const float* __restrict__ b1,
                const float* __restrict__ b2,
                const float* __restrict__ b3,
                float* __restrict__ out) {
    __shared__ __attribute__((aligned(32))) _Float16 sW[W_TOTAL];
    __shared__ __attribute__((aligned(32))) _Float16 sEf[NWAVES][16 * DH];
    __shared__ __attribute__((aligned(32))) _Float16 sEp[NWAVES][16 * DQ];
    __shared__ __attribute__((aligned(32))) _Float16 sH1[NWAVES][16 * DH];
    __shared__ __attribute__((aligned(32))) _Float16 sH2[NWAVES][16 * DQ];
    __shared__ float sCd[NWAVES][16];

    const int tid = threadIdx.x;
    // Cooperative weight staging (v8h = 16B chunks)
    for (int t = tid; t < W_TOTAL / 8; t += blockDim.x)
        ((v8h*)sW)[t] = ((const v8h*)wsh)[t];
    __syncthreads();

    const int wv = tid >> 5;
    const int lane = tid & 31;
    const int l16 = lane >> 4;
    const int ln = lane & 15;

    // Tile id -> (b, i, jt): 16 edges (i, jt*16 .. jt*16+15)
    const int tile = blockIdx.x * NWAVES + wv;
    const int b = tile / (NN_ * (NN_ / 16));
    const int rem = tile % (NN_ * (NN_ / 16));
    const int i = rem / (NN_ / 16);
    const int jt = rem % (NN_ / 16);

    // ---- ef = tanh(nproj_i + nproj_j), cd tile (wave-private LDS) ----
    _Float16* ef = sEf[wv];
    const float* npI = nproj + ((size_t)b * NN_ + i) * DH;
    for (int t = lane; t < 16 * DH; t += 32) {
        int m = t >> 6, k = t & 63;
        int j = jt * 16 + m;
        float v = tanhf(npI[k] + nproj[((size_t)b * NN_ + j) * DH + k]);
        ef[t] = (_Float16)v;
    }
    if (lane < 16) {
        int j = jt * 16 + lane;
        float c = cds[((size_t)b * NN_ + i) * NN_ + j];
        c = fminf(fmaxf(c, 0.0f), 5.0f) * 0.2f;
        sCd[wv][lane] = c;
    }
    // wave-private LDS: DS ops are in-order within a wave, no barrier needed

    // ---- ep[16x32] = ef @ W_edge[:, :64].T + b_edge ----
    _Float16* epl = sEp[wv];
    {
        const _Float16* We = sW + OFF_WE;
#pragma unroll
        for (int nt = 0; nt < 2; ++nt) {
            float bias = b_edge[nt * 16 + ln];
            v8f acc = {bias, bias, bias, bias, bias, bias, bias, bias};
            acc = wmma_f16(lds_load_A(ef, DH, ln, l16, 0),
                           lds_load_B(We, DH, nt * 16 + ln, l16, 0), acc);
            acc = wmma_f16(lds_load_A(ef, DH, ln, l16, 32),
                           lds_load_B(We, DH, nt * 16 + ln, l16, 32), acc);
#pragma unroll
            for (int v = 0; v < 8; ++v)
                epl[(v + 8 * l16) * DQ + nt * 16 + ln] = (_Float16)acc[v];
        }
    }

    // ---- h1[16x64] = relu([ef|ep|cd] @ W1.T + b1) ----
    _Float16* h1l = sH1[wv];
    {
        const _Float16* W1l = sW + OFF_W1;
#pragma unroll
        for (int nt = 0; nt < 4; ++nt) {
            int nn = nt * 16 + ln;
            float bias = b1[nn];
            float w96 = (float)W1l[nn * K1P + 96];  // cd column (k=96)
            v8f acc = {bias, bias, bias, bias, bias, bias, bias, bias};
#pragma unroll
            for (int v = 0; v < 8; ++v)
                acc[v] += sCd[wv][v + 8 * l16] * w96;  // rank-1 cd term
            acc = wmma_f16(lds_load_A(ef, DH, ln, l16, 0),
                           lds_load_B(W1l, K1P, nn, l16, 0), acc);
            acc = wmma_f16(lds_load_A(ef, DH, ln, l16, 32),
                           lds_load_B(W1l, K1P, nn, l16, 32), acc);
            acc = wmma_f16(lds_load_A(epl, DQ, ln, l16, 0),
                           lds_load_B(W1l, K1P, nn, l16, 64), acc);
#pragma unroll
            for (int v = 0; v < 8; ++v) {
                float r = fmaxf(acc[v], 0.0f);
                h1l[(v + 8 * l16) * DH + nn] = (_Float16)r;
            }
        }
    }

    // ---- h2[16x32] = relu(h1 @ W2.T + b2) ----
    _Float16* h2l = sH2[wv];
    {
        const _Float16* W2l = sW + OFF_W2;
#pragma unroll
        for (int nt = 0; nt < 2; ++nt) {
            int nn = nt * 16 + ln;
            float bias = b2[nn];
            v8f acc = {bias, bias, bias, bias, bias, bias, bias, bias};
            acc = wmma_f16(lds_load_A(h1l, DH, ln, l16, 0),
                           lds_load_B(W2l, DH, nn, l16, 0), acc);
            acc = wmma_f16(lds_load_A(h1l, DH, ln, l16, 32),
                           lds_load_B(W2l, DH, nn, l16, 32), acc);
#pragma unroll
            for (int v = 0; v < 8; ++v) {
                float r = fmaxf(acc[v], 0.0f);
                h2l[(v + 8 * l16) * DQ + nn] = (_Float16)r;
            }
        }
    }

    // ---- p[16x8] = sigmoid(h2 @ W3.T + b3) * mask, store [B,H,N,N] ----
    {
        const _Float16* W3l = sW + OFF_W3;
        float bias = (ln < HH) ? b3[ln] : 0.0f;
        v8f acc = {bias, bias, bias, bias, bias, bias, bias, bias};
        acc = wmma_f16(lds_load_A(h2l, DQ, ln, l16, 0),
                       lds_load_B(W3l, DQ, ln, l16, 0), acc);
        float mi = mask[b * NN_ + i] ? 1.0f : 0.0f;
        if (ln < HH) {
#pragma unroll
            for (int v = 0; v < 8; ++v) {
                int j = jt * 16 + 8 * l16 + v;
                float mj = mask[b * NN_ + j] ? 1.0f : 0.0f;
                acc[v] = mi * mj / (1.0f + __expf(-acc[v]));
            }
            // lane holds 8 contiguous j's for head h=ln: one 32B store
            float* dst = out + ((((size_t)b * HH + ln) * NN_ + i) * NN_) +
                         jt * 16 + 8 * l16;
            *(v8f*)dst = acc;
        }
    }
}

extern "C" void kernel_launch(void* const* d_in, const int* in_sizes, int n_in,
                              void* d_out, int out_size, void* d_ws, size_t ws_size,
                              hipStream_t stream) {
    const float* node_features = (const float*)d_in[0];
    const float* cds           = (const float*)d_in[1];
    const unsigned char* nmask = (const unsigned char*)d_in[2];
    const float* W_node        = (const float*)d_in[3];
    const float* b_node        = (const float*)d_in[4];
    const float* W_edge        = (const float*)d_in[5];
    const float* b_edge        = (const float*)d_in[6];
    const float* W1            = (const float*)d_in[7];
    const float* b1            = (const float*)d_in[8];
    const float* W2            = (const float*)d_in[9];
    const float* b2            = (const float*)d_in[10];
    const float* W3            = (const float*)d_in[11];
    const float* b3            = (const float*)d_in[12];
    float* out = (float*)d_out;

    // workspace: nproj f32 [B*N*DH], then f16 packed weights
    float* nproj = (float*)d_ws;
    _Float16* wsh = (_Float16*)(nproj + (size_t)BB * NN_ * DH);

    prep_weights_kernel<<<1, 256, 0, stream>>>(W_edge, W1, W2, W3, wsh);
    nproj_kernel<<<BB * NN_, DH, 0, stream>>>(node_features, W_node, b_node, nproj);

    const int tiles = BB * NN_ * (NN_ / 16);      // 32768 wave-tiles
    const int blocks = tiles / NWAVES;            // 8192 blocks x 128 threads
    edge_mlp_kernel<<<blocks, 32 * NWAVES, 0, stream>>>(
        nproj, cds, nmask, wsh, b_edge, b1, b2, b3, out);
}
